// VesselGAT_36077725286936
// MI455X (gfx1250) — matrix operand order repaired and checked by reading, not compiled, not run on hardware
//
#include <hip/hip_runtime.h>
#include <hip/hip_bf16.h>

typedef __attribute__((ext_vector_type(16))) _Float16 v16h;
typedef __attribute__((ext_vector_type(8)))  _Float16 v8h;
typedef __attribute__((ext_vector_type(8)))  float    v8f;

#define NEG_SLOPE 0.2f
#define TPB 256

// ---------- helpers ----------
__device__ __forceinline__ float atomicMaxF(float* addr, float val) {
    int* ai = (int*)addr;
    int old = *ai;
    while (__int_as_float(old) < val) {
        int assumed = old;
        old = atomicCAS(ai, assumed, __float_as_int(val));
        if (old == assumed) break;
    }
    return __int_as_float(old);
}

// ---------- generic fill ----------
__global__ void k_fill(float* __restrict__ p, float v, long n) {
    long i = (long)blockIdx.x * blockDim.x + threadIdx.x;
    if (i < n) p[i] = v;
}

// ---------- layer-1 GEMM (K=4, Co=256): trivially small, scalar ----------
__global__ void k_gemm_k4(const float* __restrict__ x, const float* __restrict__ W,
                          float* __restrict__ out, int N) {
    long i = (long)blockIdx.x * blockDim.x + threadIdx.x;
    if (i >= (long)N * 256) return;
    int n = (int)(i >> 8), co = (int)(i & 255);
    const float* xr = x + (size_t)n * 4;
    out[i] = xr[0] * W[co] + xr[1] * W[256 + co] + xr[2] * W[512 + co] + xr[3] * W[768 + co];
}

// ---------- weight fp32 [K,Co] -> fp16 transposed [Co,K] ----------
__global__ void k_w_to_f16t(const float* __restrict__ W, _Float16* __restrict__ Wt,
                            int K, int Co) {
    long i = (long)blockIdx.x * blockDim.x + threadIdx.x;
    if (i >= (long)K * Co) return;
    int k = (int)(i / Co), co = (int)(i % Co);
    Wt[(size_t)co * K + k] = (_Float16)W[i];
}

// ---------- WMMA GEMM, 1xNB register blocking in N ----------
// C[M,Co] = A[M,K](f16) * Bt[Co,K](f16, col-major of B).
// One wave computes a 16 x (16*NB) strip of C; the A fragment is reused by NB
// v_wmma_f32_16x16x32_f16 ops per k-step.
template <int NB>
__global__ void k_wmma_gemm(const _Float16* __restrict__ A, const _Float16* __restrict__ Bt,
                            float* __restrict__ C, int M, int K, int Co) {
    int groupsN = (Co >> 4) / NB;
    int wave = (int)((blockIdx.x * blockDim.x + threadIdx.x) >> 5);
    int lane = threadIdx.x & 31;
    int total = (M >> 4) * groupsN;
    if (wave >= total) return;               // wave-uniform: EXEC stays all-ones
    int tm = wave / groupsN, tg = wave % groupsN;
    int hi = lane >> 4;                      // 0: lanes 0-15, 1: lanes 16-31
    int l16 = lane & 15;

    // A fragment (16x32 f16): lane<16 -> row l16, K {0..7,16..23}; lane>=16 -> K {8..15,24..31}
    const _Float16* arow = A + (size_t)(tm * 16 + l16) * K + hi * 8;
    // B fragments (32x16 f16 from Bt): lane<16 col l16 K {0..7,16..23}; lane>=16 K {8..15,24..31}
    const _Float16* bbase = Bt + (size_t)(tg * NB * 16 + l16) * K + hi * 8;

    v8f acc[NB];
    #pragma unroll
    for (int j = 0; j < NB; ++j) acc[j] = (v8f){};

    for (int k0 = 0; k0 < K; k0 += 32) {
        __builtin_prefetch(arow + k0 + 64, 0, 1);
        v8h a_lo = *(const v8h*)(arow + k0);
        v8h a_hi = *(const v8h*)(arow + k0 + 16);
        v16h a;
        #pragma unroll
        for (int i = 0; i < 8; ++i) { a[i] = a_lo[i]; a[i + 8] = a_hi[i]; }
        #pragma unroll
        for (int j = 0; j < NB; ++j) {
            const _Float16* brow = bbase + (size_t)j * 16 * K;
            v8h b_lo = *(const v8h*)(brow + k0);
            v8h b_hi = *(const v8h*)(brow + k0 + 16);
            v16h b;
            #pragma unroll
            for (int i = 0; i < 8; ++i) { b[i] = b_lo[i]; b[i + 8] = b_hi[i]; }
            acc[j] = __builtin_amdgcn_wmma_f32_16x16x32_f16(
                /*neg_a=*/false, a, /*neg_b=*/false, b,
                /*c_mod=*/(short)0, acc[j], /*reuse_a=*/false, /*reuse_b=*/false);
        }
    }
    // D layout: VGPR r -> M = r + 8*hi, N = l16
    #pragma unroll
    for (int j = 0; j < NB; ++j) {
        float* crow = C + (size_t)(tm * 16 + hi * 8) * Co + (tg * NB + j) * 16 + l16;
        #pragma unroll
        for (int r = 0; r < 8; ++r) crow[(size_t)r * Co] = acc[j][r];
    }
}

// ---------- edge pass 1: attention logits + segment max over dst ----------
__global__ void k_edge_logits(const float* __restrict__ xl, const float* __restrict__ xr,
                              const float* __restrict__ att, const long long* __restrict__ ei,
                              int E, int N, float* __restrict__ ebuf, float* __restrict__ emax,
                              int H, int C) {
    long ET = (long)E + N;
    long i = (long)blockIdx.x * blockDim.x + threadIdx.x;
    if (i >= ET * H) return;
    long e = i / H; int h = (int)(i % H);
    int s, d;
    if (e < E) { s = (int)ei[e]; d = (int)ei[(long)E + e]; }
    else       { s = d = (int)(e - E); }
    const float4* pl = (const float4*)(xl + (size_t)s * H * C + (size_t)h * C);
    const float4* pr = (const float4*)(xr + (size_t)d * H * C + (size_t)h * C);
    const float4* pa = (const float4*)(att + (size_t)h * C);
    float acc = 0.f;
    int C4 = C >> 2;
    for (int c = 0; c < C4; ++c) {
        float4 l = pl[c], r = pr[c], a = pa[c];
        float m0 = l.x + r.x; m0 = m0 > 0.f ? m0 : NEG_SLOPE * m0;
        float m1 = l.y + r.y; m1 = m1 > 0.f ? m1 : NEG_SLOPE * m1;
        float m2 = l.z + r.z; m2 = m2 > 0.f ? m2 : NEG_SLOPE * m2;
        float m3 = l.w + r.w; m3 = m3 > 0.f ? m3 : NEG_SLOPE * m3;
        acc += m0 * a.x + m1 * a.y + m2 * a.z + m3 * a.w;
    }
    ebuf[i] = acc;
    atomicMaxF(&emax[(size_t)d * H + h], acc);
}

// ---------- edge pass 2: exp(e - max) + segment sum ----------
__global__ void k_edge_exp(float* __restrict__ ebuf, const float* __restrict__ emax,
                           float* __restrict__ denom, const long long* __restrict__ ei,
                           int E, int N, int H) {
    long ET = (long)E + N;
    long i = (long)blockIdx.x * blockDim.x + threadIdx.x;
    if (i >= ET * H) return;
    long e = i / H; int h = (int)(i % H);
    int d = (e < E) ? (int)ei[(long)E + e] : (int)(e - E);
    float v = __expf(ebuf[i] - emax[(size_t)d * H + h]);
    ebuf[i] = v;
    atomicAdd(&denom[(size_t)d * H + h], v);
}

// ---------- edge pass 3: alpha-weighted scatter-add of xl[src] ----------
__global__ void k_edge_agg(const float* __restrict__ xl, const float* __restrict__ ebuf,
                           const float* __restrict__ denom, float* __restrict__ agg,
                           const long long* __restrict__ ei, int E, int N, int H, int C) {
    long ET = (long)E + N;
    int C4 = C >> 2, HC4 = H * C4;
    long i = (long)blockIdx.x * blockDim.x + threadIdx.x;
    if (i >= ET * HC4) return;
    long e = i / HC4; int t = (int)(i % HC4);
    int h = t / C4, c = (t % C4) * 4;
    int s, d;
    if (e < E) { s = (int)ei[e]; d = (int)ei[(long)E + e]; }
    else       { s = d = (int)(e - E); }
    float alpha = ebuf[e * H + h] / (denom[(size_t)d * H + h] + 1e-16f);
    float4 v = *(const float4*)(xl + (size_t)s * H * C + (size_t)h * C + c);
    float* pd = agg + (size_t)d * H * C + (size_t)h * C + c;
    atomicAdd(pd + 0, alpha * v.x);
    atomicAdd(pd + 1, alpha * v.y);
    atomicAdd(pd + 2, alpha * v.z);
    atomicAdd(pd + 3, alpha * v.w);
}

// ---------- bias + ELU + fp16 copy for the next WMMA GEMM ----------
__global__ void k_bias_elu_f16(const float* __restrict__ agg, const float* __restrict__ bias,
                               float* __restrict__ h, _Float16* __restrict__ h16, long n) {
    long i = (long)blockIdx.x * blockDim.x + threadIdx.x;
    if (i >= n) return;
    float v = agg[i] + bias[i & 255];
    v = v > 0.f ? v : expm1f(v);
    h[i] = v;
    h16[i] = (_Float16)v;
}

// ---------- final layer: bias only (heads=1, mean over 1 head = identity) ----------
__global__ void k_bias32(const float* __restrict__ agg, const float* __restrict__ bias,
                         float* __restrict__ z, long n) {
    long i = (long)blockIdx.x * blockDim.x + threadIdx.x;
    if (i >= n) return;
    z[i] = agg[i] + bias[i & 31];
}

// ---------- per-graph pooling ----------
__global__ void k_pool_accum(const float* __restrict__ z, const long long* __restrict__ batch,
                             float* __restrict__ sums, float* __restrict__ maxs,
                             float* __restrict__ counts, int N) {
    long i = (long)blockIdx.x * blockDim.x + threadIdx.x;
    if (i >= (long)N * 32) return;
    int n = (int)(i >> 5), c = (int)(i & 31);
    int b = (int)batch[n];
    float v = z[i];
    atomicAdd(&sums[b * 32 + c], v);
    atomicMaxF(&maxs[b * 32 + c], v);
    if (c == 0) atomicAdd(&counts[b], 1.0f);
}

__global__ void k_pool_final(const float* __restrict__ sums, const float* __restrict__ maxs,
                             const float* __restrict__ counts, float* __restrict__ out) {
    int i = blockIdx.x * blockDim.x + threadIdx.x;
    if (i >= 512) return;
    int b = i >> 6, c = i & 63;
    out[i] = (c < 32) ? sums[b * 32 + c] / counts[b] : maxs[b * 32 + (c - 32)];
}

// ---------- host ----------
extern "C" void kernel_launch(void* const* d_in, const int* in_sizes, int n_in,
                              void* d_out, int out_size, void* d_ws, size_t ws_size,
                              hipStream_t stream) {
    const float*     x     = (const float*)d_in[0];
    const long long* ei    = (const long long*)d_in[1];
    const long long* batch = (const long long*)d_in[2];
    const float* W1l = (const float*)d_in[3];  const float* W1r = (const float*)d_in[4];
    const float* a1  = (const float*)d_in[5];  const float* b1  = (const float*)d_in[6];
    const float* W2l = (const float*)d_in[7];  const float* W2r = (const float*)d_in[8];
    const float* a2  = (const float*)d_in[9];  const float* b2  = (const float*)d_in[10];
    const float* W3l = (const float*)d_in[11]; const float* W3r = (const float*)d_in[12];
    const float* a3  = (const float*)d_in[13]; const float* b3  = (const float*)d_in[14];

    const int N = in_sizes[0] / 4;
    const int E = in_sizes[1] / 2;
    const long ET = (long)E + N;

    // workspace carve-out (256B aligned)
    char* ws = (char*)d_ws;
    auto carve = [&](size_t bytes) -> char* {
        char* p = ws; ws += (bytes + 255) & ~(size_t)255; return p;
    };
    float*    h_cur = (float*)carve((size_t)N * 256 * 4);
    float*    xl    = (float*)carve((size_t)N * 256 * 4);
    float*    xr    = (float*)carve((size_t)N * 256 * 4);
    float*    agg   = (float*)carve((size_t)N * 256 * 4);
    _Float16* h16   = (_Float16*)carve((size_t)N * 256 * 2);
    float*    ebuf  = (float*)carve((size_t)ET * 4 * 4);
    float*    emax  = (float*)carve((size_t)N * 4 * 4);
    float*    denom = (float*)carve((size_t)N * 4 * 4);
    _Float16* wtl   = (_Float16*)carve((size_t)256 * 256 * 2);
    _Float16* wtr   = (_Float16*)carve((size_t)256 * 256 * 2);
    float*    z     = (float*)carve((size_t)N * 32 * 4);
    float*    psum  = (float*)carve(8 * 32 * 4);
    float*    pmax  = (float*)carve(8 * 32 * 4);
    float*    pcnt  = (float*)carve(8 * 4);

    auto nb = [](long n) { return (unsigned)((n + TPB - 1) / TPB); };
    const float NINF = -__builtin_huge_valf();

    // ===== Layer 1 (K=4, heads=4, C=64) =====
    k_gemm_k4<<<nb((long)N * 256), TPB, 0, stream>>>(x, W1l, xl, N);
    k_gemm_k4<<<nb((long)N * 256), TPB, 0, stream>>>(x, W1r, xr, N);
    k_fill<<<nb((long)N * 4), TPB, 0, stream>>>(emax, NINF, (long)N * 4);
    k_fill<<<nb((long)N * 4), TPB, 0, stream>>>(denom, 0.f, (long)N * 4);
    k_fill<<<nb((long)N * 256), TPB, 0, stream>>>(agg, 0.f, (long)N * 256);
    k_edge_logits<<<nb(ET * 4), TPB, 0, stream>>>(xl, xr, a1, ei, E, N, ebuf, emax, 4, 64);
    k_edge_exp<<<nb(ET * 4), TPB, 0, stream>>>(ebuf, emax, denom, ei, E, N, 4);
    k_edge_agg<<<nb(ET * 64), TPB, 0, stream>>>(xl, ebuf, denom, agg, ei, E, N, 4, 64);
    k_bias_elu_f16<<<nb((long)N * 256), TPB, 0, stream>>>(agg, b1, h_cur, h16, (long)N * 256);

    // ===== Layer 2 (K=256, heads=4, C=64) — WMMA GEMMs, 1x4 N-blocking =====
    k_w_to_f16t<<<nb(256 * 256), TPB, 0, stream>>>(W2l, wtl, 256, 256);
    k_w_to_f16t<<<nb(256 * 256), TPB, 0, stream>>>(W2r, wtr, 256, 256);
    {
        long waves = (long)(N / 16) * ((256 / 16) / 4);
        k_wmma_gemm<4><<<nb(waves * 32), TPB, 0, stream>>>(h16, wtl, xl, N, 256, 256);
        k_wmma_gemm<4><<<nb(waves * 32), TPB, 0, stream>>>(h16, wtr, xr, N, 256, 256);
    }
    k_fill<<<nb((long)N * 4), TPB, 0, stream>>>(emax, NINF, (long)N * 4);
    k_fill<<<nb((long)N * 4), TPB, 0, stream>>>(denom, 0.f, (long)N * 4);
    k_fill<<<nb((long)N * 256), TPB, 0, stream>>>(agg, 0.f, (long)N * 256);
    k_edge_logits<<<nb(ET * 4), TPB, 0, stream>>>(xl, xr, a2, ei, E, N, ebuf, emax, 4, 64);
    k_edge_exp<<<nb(ET * 4), TPB, 0, stream>>>(ebuf, emax, denom, ei, E, N, 4);
    k_edge_agg<<<nb(ET * 64), TPB, 0, stream>>>(xl, ebuf, denom, agg, ei, E, N, 4, 64);
    k_bias_elu_f16<<<nb((long)N * 256), TPB, 0, stream>>>(agg, b2, h_cur, h16, (long)N * 256);

    // ===== Layer 3 (K=256, heads=1, C=32) — WMMA GEMMs, 1x2 N-blocking =====
    k_w_to_f16t<<<nb(256 * 32), TPB, 0, stream>>>(W3l, wtl, 256, 32);
    k_w_to_f16t<<<nb(256 * 32), TPB, 0, stream>>>(W3r, wtr, 256, 32);
    {
        long waves = (long)(N / 16) * ((32 / 16) / 2);
        k_wmma_gemm<2><<<nb(waves * 32), TPB, 0, stream>>>(h16, wtl, xl, N, 256, 32);
        k_wmma_gemm<2><<<nb(waves * 32), TPB, 0, stream>>>(h16, wtr, xr, N, 256, 32);
    }
    k_fill<<<nb((long)N), TPB, 0, stream>>>(emax, NINF, (long)N);
    k_fill<<<nb((long)N), TPB, 0, stream>>>(denom, 0.f, (long)N);
    k_fill<<<nb((long)N * 32), TPB, 0, stream>>>(agg, 0.f, (long)N * 32);
    k_edge_logits<<<nb(ET), TPB, 0, stream>>>(xl, xr, a3, ei, E, N, ebuf, emax, 1, 32);
    k_edge_exp<<<nb(ET), TPB, 0, stream>>>(ebuf, emax, denom, ei, E, N, 1);
    k_edge_agg<<<nb(ET * 8), TPB, 0, stream>>>(xl, ebuf, denom, agg, ei, E, N, 1, 32);
    k_bias32<<<nb((long)N * 32), TPB, 0, stream>>>(agg, b3, z, (long)N * 32);

    // ===== per-graph mean/max pooling -> [8, 64] =====
    k_fill<<<1, TPB, 0, stream>>>(psum, 0.f, 256);
    k_fill<<<1, TPB, 0, stream>>>(pmax, NINF, 256);
    k_fill<<<1, TPB, 0, stream>>>(pcnt, 0.f, 8);
    k_pool_accum<<<nb((long)N * 32), TPB, 0, stream>>>(z, batch, psum, pmax, pcnt, N);
    k_pool_final<<<2, TPB, 0, stream>>>(psum, pmax, pcnt, (float*)d_out);
}
